// CAL_51015621542567
// MI455X (gfx1250) — matrix operand order repaired and checked by reading, not compiled, not run on hardware
//
#include <hip/hip_runtime.h>

// MI455X / gfx1250, wave32. Matrix math via v_wmma_f32_16x16x32_bf16.
// Workgroup-shared GEMM operands staged to LDS with global_load_async_to_lds_b128
// (ASYNCcnt) in a fragment-linear layout -> conflict-free ds_load_b128 in the loop.
#define B_ 32
#define D_ 768
#define T_ 768
#define L_ 1024
#define EPSF 1e-5f

typedef __attribute__((ext_vector_type(16))) __bf16 v16bf;
typedef __attribute__((ext_vector_type(8)))  float  v8f;
typedef __attribute__((ext_vector_type(8)))  unsigned short ush8;
typedef unsigned short u16;

union Frag { v16bf bf; ush8 u[2]; };

__device__ __forceinline__ u16 f2bf(float f) {
  unsigned x = __float_as_uint(f);
  unsigned r = x + 0x7FFFu + ((x >> 16) & 1u);   // round-to-nearest-even
  return (u16)(r >> 16);
}

// ---- CDNA5 async global->LDS copy (ASYNCcnt). 16B per lane, LDS scatter via VDST.
__device__ __forceinline__ void async_cp16(unsigned lds_off, const u16* gaddr) {
  asm volatile("global_load_async_to_lds_b128 %0, %1, off"
               :: "v"(lds_off), "v"((unsigned long long)(size_t)gaddr)
               : "memory");
}
__device__ __forceinline__ void async_wait0() {
  asm volatile("s_wait_asynccnt 0x0" ::: "memory");
}

// A-matrix 16x32 bf16 (MxK) frag, row-major global memory, leading dim `ld`.
// lanes 0-15: M=lane, K=k0+[0..7],k0+[16..23]; lanes 16-31: M=lane-16, K=k0+[8..15],[24..31].
__device__ __forceinline__ v16bf load_a(const u16* base, int ld, int k0, int lane) {
  int row  = lane & 15;
  int koff = k0 + ((lane & 16) ? 8 : 0);
  const u16* p = base + (size_t)row * ld + koff;
  Frag f;
  f.u[0] = *(const ush8*)(p);
  f.u[1] = *(const ush8*)(p + 16);
  return f.bf;
}

// Frag from LDS fragment-linear layout: step*1024B + lane*32B (u16 units: step*512+lane*16).
__device__ __forceinline__ v16bf frag_from_lds(const u16* smem, int step, int lane) {
  const u16* p = smem + (size_t)step * 512 + lane * 16;
  Frag f;
  f.u[0] = *(const ush8*)(p);       // ds_load_b128, conflict-free
  f.u[1] = *(const ush8*)(p + 8);
  return f.bf;
}

// Stage A-operand tile (16 rows x 32*ksteps K) into fragment-linear LDS.
// chunk c (16B): step=c>>6, slot=(c&63)>>1, half=c&1;
// A lane layout: koff = step*32 + (slot&16 ? 16:0) + half*8.
__device__ __forceinline__ void stage_a_tile(unsigned sbase, const u16* src, int ld,
                                             int ksteps, int tid) {
  int nchunks = ksteps * 64;
  for (int c = tid; c < nchunks; c += 256) {
    int step = c >> 6, m = c & 63, slot = m >> 1, half = m & 1;
    int row  = slot & 15;
    int koff = step * 32 + ((slot & 16) ? 16 : 0) + half * 8;
    async_cp16(sbase + (unsigned)c * 16u, src + (size_t)row * ld + koff);
  }
}

// Stage B-operand tile (16 cols x 32*ksteps K, each col K-contiguous in memory).
// B lane layout: koff = step*32 + (slot&16 ? 8:0) + half*16.
__device__ __forceinline__ void stage_b_tile(unsigned sbase, const u16* src, int ld,
                                             int ksteps, int tid) {
  int nchunks = ksteps * 64;
  for (int c = tid; c < nchunks; c += 256) {
    int step = c >> 6, m = c & 63, slot = m >> 1, half = m & 1;
    int col  = slot & 15;
    int koff = step * 32 + ((slot & 16) ? 8 : 0) + half * 16;
    async_cp16(sbase + (unsigned)c * 16u, src + (size_t)col * ld + koff);
  }
}

#define WMMA_BF16(a, b, c) \
  __builtin_amdgcn_wmma_f32_16x16x32_bf16(false, (a), false, (b), (short)0, (c), false, false)

// ---------------- K1a: f32 -> bf16 convert ----------------
__global__ __launch_bounds__(256) void k_cvt(const float* __restrict__ in,
                                             u16* __restrict__ out, int n) {
  int i = blockIdx.x * blockDim.x + threadIdx.x;
  int stride = gridDim.x * blockDim.x;
  for (; i < n; i += stride) out[i] = f2bf(in[i]);
}

// ---------------- K1b: transpose y[b,t,l] -> yT[b,l,t] (bf16) ----------------
__global__ __launch_bounds__(256) void k_transpose(const float* __restrict__ y,
                                                   u16* __restrict__ yT) {
  __shared__ u16 tile[32][33];
  int b = blockIdx.z;
  int t0 = blockIdx.y * 32, l0 = blockIdx.x * 32;
  int tx = threadIdx.x, ty = threadIdx.y;   // 32 x 8
  const float* yb = y + (size_t)b * T_ * L_;
  u16* yTb = yT + (size_t)b * L_ * T_;
  for (int i = 0; i < 32; i += 8)
    tile[ty + i][tx] = f2bf(yb[(size_t)(t0 + ty + i) * L_ + l0 + tx]);
  __syncthreads();
  for (int i = 0; i < 32; i += 8)
    yTb[(size_t)(l0 + ty + i) * T_ + t0 + tx] = tile[tx][ty + i];
}

// ---------------- K2: S = X.Y^T (K=L=1024) + row softmax -> att bf16 ----------
// grid (D/16, B), block 256 (8 waves). A tile shared by all waves -> async->LDS.
__global__ __launch_bounds__(256) void k_attn(const u16* __restrict__ xb,
                                              const u16* __restrict__ yb,
                                              u16* __restrict__ att) {
  __shared__ u16 sA[32 * 512];          // 32KB fragment-linear A tile
  __shared__ float redmax[8][16];
  __shared__ float redsum[8][16];
  int b = blockIdx.y;
  int dt = blockIdx.x;
  int tid = threadIdx.x;
  int wave = tid >> 5, lane = tid & 31;
  const u16* A  = xb + (size_t)b * D_ * L_ + (size_t)dt * 16 * L_;
  const u16* Yb = yb + (size_t)b * T_ * L_;

  stage_a_tile((unsigned)(size_t)(void*)sA, A, L_, 32, tid);
  async_wait0();
  __syncthreads();

  v8f acc[6];
  {
    v8f z = {0.f,0.f,0.f,0.f,0.f,0.f,0.f,0.f};
    for (int j = 0; j < 6; ++j) acc[j] = z;
  }
  for (int step = 0; step < 32; ++step) {
    v16bf af = frag_from_lds(sA, step, lane);
    int k0 = step * 32;
    for (int j = 0; j < 6; ++j) {
      int tt = (wave * 6 + j) * 16;
      // B frag straight from global (distinct per wave; Y[b] is L2-resident)
      int col  = lane & 15;
      int koff = k0 + ((lane & 16) ? 16 : 0);
      const u16* p = Yb + (size_t)(tt + col) * L_ + koff;
      Frag f;
      f.u[0] = *(const ush8*)(p);
      f.u[1] = *(const ush8*)(p + 8);
      acc[j] = WMMA_BF16(af, f.bf, acc[j]);
    }
  }

  // softmax over t per d-row. C layout: reg r -> M=r (lanes 0-15) / r+8 (16-31), N=lane&15.
  int halfoff = (lane & 16) ? 8 : 0;
  float gmax[8], ginv[8];
  for (int r = 0; r < 8; ++r) {
    float m = -3.4e38f;
    for (int j = 0; j < 6; ++j) m = fmaxf(m, acc[j][r]);
    for (int s = 1; s < 16; s <<= 1) m = fmaxf(m, __shfl_xor(m, s, 32));
    if ((lane & 15) == 0) redmax[wave][r + halfoff] = m;
  }
  __syncthreads();
  for (int r = 0; r < 8; ++r) {
    float m = -3.4e38f;
    for (int w = 0; w < 8; ++w) m = fmaxf(m, redmax[w][r + halfoff]);
    gmax[r] = m;
  }
  for (int r = 0; r < 8; ++r) {
    float s = 0.f;
    for (int j = 0; j < 6; ++j) {
      float p = __expf(acc[j][r] - gmax[r]);
      acc[j][r] = p;
      s += p;
    }
    for (int t = 1; t < 16; t <<= 1) s += __shfl_xor(s, t, 32);
    if ((lane & 15) == 0) redsum[wave][r + halfoff] = s;
  }
  __syncthreads();
  for (int r = 0; r < 8; ++r) {
    float s = 0.f;
    for (int w = 0; w < 8; ++w) s += redsum[w][r + halfoff];
    ginv[r] = 1.0f / s;
  }
  for (int r = 0; r < 8; ++r) {
    int d = dt * 16 + r + halfoff;
    for (int j = 0; j < 6; ++j) {
      int t = (wave * 6 + j) * 16 + (lane & 15);
      att[(size_t)b * D_ * T_ + (size_t)d * T_ + t] = f2bf(acc[j][r] * ginv[r]);
    }
  }
}

// ---------------- K3: x_res = att.Y (K=T=768), h = x + gav*x_res, pre-LN over D,
// ReLU, BN partials, store hT[b,l,d] bf16 via LDS-coalesced flush. -------------
// grid (L/16, B), block 256. B tile (yT slice) shared by all waves -> async->LDS.
__global__ __launch_bounds__(256) void k_av_ln(const u16* __restrict__ att,
                                               const u16* __restrict__ yT,
                                               const float* __restrict__ x,
                                               const float* __restrict__ gate_av,
                                               const float* __restrict__ lnw,
                                               const float* __restrict__ lnb,
                                               u16* __restrict__ hT,
                                               float* __restrict__ bnsum,
                                               float* __restrict__ bnsq) {
  __shared__ u16 smem[16 * 776];        // 24.25KB: B tile (24KB), then reused as hT tile
  __shared__ float lnS[16], lnQ[16];
  int b = blockIdx.y;
  int l0 = blockIdx.x * 16;
  int tid = threadIdx.x;
  int wave = tid >> 5, lane = tid & 31;
  const u16* Bm = yT + (size_t)b * L_ * T_ + (size_t)l0 * T_;
  const u16* Ab = att + (size_t)b * D_ * T_;

  stage_b_tile((unsigned)(size_t)(void*)smem, Bm, T_, 24, tid);
  async_wait0();
  __syncthreads();

  v8f acc[6];
  {
    v8f z = {0.f,0.f,0.f,0.f,0.f,0.f,0.f,0.f};
    for (int j = 0; j < 6; ++j) acc[j] = z;
  }
  for (int step = 0; step < 24; ++step) {
    v16bf bfr = frag_from_lds(smem, step, lane);
    int k0 = step * 32;
    for (int j = 0; j < 6; ++j) {
      int dtile = wave * 6 + j;
      v16bf af = load_a(Ab + (size_t)dtile * 16 * T_, T_, k0, lane);
      acc[j] = WMMA_BF16(af, bfr, acc[j]);
    }
  }

  int halfoff = (lane & 16) ? 8 : 0;
  int l = l0 + (lane & 15);
  float gav = gate_av[0];
  float ps = 0.f, pq = 0.f;
  for (int j = 0; j < 6; ++j)
    for (int r = 0; r < 8; ++r) {
      int d = (wave * 6 + j) * 16 + r + halfoff;
      float h = x[(size_t)b * D_ * L_ + (size_t)d * L_ + l] + gav * acc[j][r];
      acc[j][r] = h;
      ps += h; pq += h * h;
    }
  // LN over D: all 768 d live in this workgroup.
  if (tid < 16) { lnS[tid] = 0.f; lnQ[tid] = 0.f; }
  __syncthreads();                       // also ends all reads of B tile in smem
  atomicAdd(&lnS[lane & 15], ps);
  atomicAdd(&lnQ[lane & 15], pq);
  __syncthreads();
  float mu  = lnS[lane & 15] * (1.f / D_);
  float var = lnQ[lane & 15] * (1.f / D_) - mu * mu;
  float rs  = rsqrtf(var + EPSF);
  for (int j = 0; j < 6; ++j)
    for (int r = 0; r < 8; ++r) {
      int d = (wave * 6 + j) * 16 + r + halfoff;
      float hn = (acc[j][r] - mu) * rs * lnw[d] + lnb[d];
      float hr = fmaxf(hn, 0.f);
      // bank-safe LDS transpose tile (row stride 776 u16 -> lane bank stride 4)
      smem[(size_t)(lane & 15) * 776 + d] = f2bf(hr);
      float s = hr, q = hr * hr;
      for (int m = 1; m < 16; m <<= 1) { s += __shfl_xor(s, m, 32); q += __shfl_xor(q, m, 32); }
      if ((lane & 15) == 0) { atomicAdd(&bnsum[d], s); atomicAdd(&bnsq[d], q); }
    }
  __syncthreads();
  // coalesced flush: 16 rows x 96 16B-chunks
  u16* hTb = hT + (size_t)b * L_ * D_;
  for (int c = tid; c < 1536; c += 256) {
    int row = c / 96;
    int off = c - row * 96;
    ush8 v = *(const ush8*)(smem + (size_t)row * 776 + off * 8);
    *(ush8*)(hTb + (size_t)(l0 + row) * D_ + off * 8) = v;
  }
}

// ---------------- K4: BN coefficients a_d, c_d ----------------
__global__ __launch_bounds__(256) void k_bncoef(const float* __restrict__ bnsum,
                                                const float* __restrict__ bnsq,
                                                const float* __restrict__ gamma,
                                                const float* __restrict__ beta,
                                                float* __restrict__ ca,
                                                float* __restrict__ cc) {
  int d = blockIdx.x * blockDim.x + threadIdx.x;
  if (d >= D_) return;
  float n = (float)(B_ * L_);
  float mu = bnsum[d] / n;
  float var = bnsq[d] / n - mu * mu;
  float a = gamma[d] * rsqrtf(var + EPSF);
  ca[d] = a;
  cc[d] = beta[d] - mu * a;
}

// ---------------- K5: fold BN into weights: wf = bf16(w*a), bias_o = sum w*c ----
__global__ __launch_bounds__(256) void k_fold(const float* __restrict__ w,
                                              const float* __restrict__ ca,
                                              const float* __restrict__ cc,
                                              u16* __restrict__ wf,
                                              float* __restrict__ bias) {
  int o = blockIdx.x;
  float local = 0.f;
  for (int d = threadIdx.x; d < D_; d += 256) {
    float ww = w[(size_t)o * D_ + d];
    wf[(size_t)o * D_ + d] = f2bf(ww * ca[d]);
    local += ww * cc[d];
  }
  __shared__ float red[256];
  red[threadIdx.x] = local;
  __syncthreads();
  for (int s = 128; s > 0; s >>= 1) {
    if (threadIdx.x < s) red[threadIdx.x] += red[threadIdx.x + s];
    __syncthreads();
  }
  if (threadIdx.x == 0) bias[o] = red[0];
}

// ---------------- K6: out = gate * postLN( wf.h + bias ). grid (L/16, B). ------
__global__ __launch_bounds__(256) void k_mlp_ln(const u16* __restrict__ wf,
                                                const u16* __restrict__ hT,
                                                const float* __restrict__ bias,
                                                const float* __restrict__ lnpw,
                                                const float* __restrict__ lnpb,
                                                const float* __restrict__ gate,
                                                float* __restrict__ out) {
  __shared__ u16 sB[24 * 512];          // 24KB fragment-linear B tile
  __shared__ float lnS[16], lnQ[16];
  int b = blockIdx.y;
  int l0 = blockIdx.x * 16;
  int tid = threadIdx.x;
  int wave = tid >> 5, lane = tid & 31;
  const u16* Bm = hT + (size_t)b * L_ * D_ + (size_t)l0 * D_;

  stage_b_tile((unsigned)(size_t)(void*)sB, Bm, D_, 24, tid);
  async_wait0();
  __syncthreads();

  v8f acc[6];
  {
    v8f z = {0.f,0.f,0.f,0.f,0.f,0.f,0.f,0.f};
    for (int j = 0; j < 6; ++j) acc[j] = z;
  }
  for (int step = 0; step < 24; ++step) {
    v16bf bfr = frag_from_lds(sB, step, lane);
    int k0 = step * 32;
    for (int j = 0; j < 6; ++j) {
      v16bf af = load_a(wf + (size_t)(wave * 6 + j) * 16 * D_, D_, k0, lane);
      acc[j] = WMMA_BF16(af, bfr, acc[j]);
    }
  }

  int halfoff = (lane & 16) ? 8 : 0;
  int l = l0 + (lane & 15);
  float ps = 0.f, pq = 0.f;
  for (int j = 0; j < 6; ++j)
    for (int r = 0; r < 8; ++r) {
      int o = (wave * 6 + j) * 16 + r + halfoff;
      float g = acc[j][r] + bias[o];
      acc[j][r] = g;
      ps += g; pq += g * g;
    }
  if (tid < 16) { lnS[tid] = 0.f; lnQ[tid] = 0.f; }
  __syncthreads();
  atomicAdd(&lnS[lane & 15], ps);
  atomicAdd(&lnQ[lane & 15], pq);
  __syncthreads();
  float mu  = lnS[lane & 15] * (1.f / D_);
  float var = lnQ[lane & 15] * (1.f / D_) - mu * mu;
  float rs  = rsqrtf(var + EPSF);
  float gt = gate[0];
  for (int j = 0; j < 6; ++j)
    for (int r = 0; r < 8; ++r) {
      int o = (wave * 6 + j) * 16 + r + halfoff;
      float v = ((acc[j][r] - mu) * rs * lnpw[o] + lnpb[o]) * gt;
      out[(size_t)b * D_ * L_ + (size_t)o * L_ + l] = v;   // coalesced: lanes -> consecutive l
    }
}

extern "C" void kernel_launch(void* const* d_in, const int* in_sizes, int n_in,
                              void* d_out, int out_size, void* d_ws, size_t ws_size,
                              hipStream_t stream) {
  const float* x       = (const float*)d_in[0];
  const float* y       = (const float*)d_in[1];
  const float* gate_av = (const float*)d_in[2];
  const float* gate    = (const float*)d_in[3];
  const float* lnw     = (const float*)d_in[4];
  const float* lnb     = (const float*)d_in[5];
  const float* gamma   = (const float*)d_in[6];
  const float* beta    = (const float*)d_in[7];
  const float* mlpw    = (const float*)d_in[8];
  const float* lnpw    = (const float*)d_in[9];
  const float* lnpb    = (const float*)d_in[10];
  float* out = (float*)d_out;

  const size_t NX = (size_t)B_ * D_ * L_;    // 25,165,824 elements
  char* w = (char*)d_ws;
  u16* xb  = (u16*)w;  w += NX * 2;
  u16* yb  = (u16*)w;  w += NX * 2;
  u16* yT  = (u16*)w;  w += NX * 2;
  u16* att = (u16*)w;  w += (size_t)B_ * D_ * T_ * 2;
  u16* hT  = (u16*)w;  w += NX * 2;
  u16* wf  = (u16*)w;  w += (size_t)D_ * D_ * 2;
  float* bias  = (float*)w; w += D_ * 4;
  float* bnsum = (float*)w; w += D_ * 4;
  float* bnsq  = (float*)w; w += D_ * 4;
  float* ca    = (float*)w; w += D_ * 4;
  float* cc    = (float*)w; w += D_ * 4;

  hipMemsetAsync(bnsum, 0, (size_t)D_ * 4 * 2, stream);  // bnsum + bnsq (contiguous)

  k_cvt<<<8192, 256, 0, stream>>>(x, xb, (int)NX);
  k_cvt<<<8192, 256, 0, stream>>>(y, yb, (int)NX);
  k_transpose<<<dim3(L_ / 32, T_ / 32, B_), dim3(32, 8), 0, stream>>>(y, yT);
  k_attn<<<dim3(D_ / 16, B_), 256, 0, stream>>>(xb, yb, att);
  k_av_ln<<<dim3(L_ / 16, B_), 256, 0, stream>>>(att, yT, x, gate_av, lnw, lnb, hT, bnsum, bnsq);
  k_bncoef<<<3, 256, 0, stream>>>(bnsum, bnsq, gamma, beta, ca, cc);
  k_fold<<<D_, 256, 0, stream>>>(mlpw, ca, cc, wf, bias);
  k_mlp_ln<<<dim3(L_ / 16, B_), 256, 0, stream>>>(wf, hT, bias, lnpw, lnpb, gate, out);
}
